// LanguageAdapter_15006615733345
// MI455X (gfx1250) — matrix operand order repaired and checked by reading, not compiled, not run on hardware
//
#include <hip/hip_runtime.h>
#include <hip/hip_bf16.h>

// ---------------------------------------------------------------------------
// LanguageAdapter fused kernel for MI455X (gfx1250), wave32 + bf16 WMMA.
//   out = LayerNorm(hidden + relu(hidden @ W1[lang] + b1[lang]) @ W2[lang]
//                   + b2[lang]) * gamma[lang] + beta[lang]
// Shapes: B=32, S=2048, H=1024, A=256, L=10 (fp32 in/out).
// - BM=64 rows/block, 512 threads (16 waves, 4/SIMD): low register pressure
//   so batched B-fragments stay in distinct regs; good latency hiding.
// - Hidden/residual tile kept in bf16; LN math in f32.
// - WMMA fragments: 2x ds_load_b128 each (A: row-major bf16 tiles, B:
//   K-contiguous transposed weight stage).
// - f32->bf16 packing: single v_cvt_pk_bf16_f32 per pair (inline asm).
// - Weight staging: register-pipelined global b128 loads + LDS double buffer
//   (one barrier per K-chunk).
// ---------------------------------------------------------------------------

#define BATCH 32
#define SEQ   2048
#define HD    1024
#define AD    256
#define BM    64          // rows per workgroup tile
#define BKK   32          // K step per WMMA chunk (bf16 WMMA is 16x16x32)
#define NT    512         // threads per block (16 waves)

// LDS strides (elements), padded to spread banks
#define ASTR   1032                        // bf16 hidden/x tile row stride
#define HSTR   264                         // bf16 h-tile row stride
#define KSTR   40                          // transposed weight stage: [N][KSTR]

#define ABF_BYTES    (BM * ASTR * 2)       // 132096
#define HTILE_BYTES  (BM * HSTR * 2)       // 33792
#define BSTG_BYTES   (AD * KSTR * 2)       // 20480 per buffer (max N=256)
#define BSTG_ELEMS   (AD * KSTR)
#define SMEM_BYTES   (ABF_BYTES + HTILE_BYTES + 2 * BSTG_BYTES)   // 206848

typedef __bf16 bf16_t;
typedef __attribute__((ext_vector_type(16))) __bf16 v16bf;
typedef __attribute__((ext_vector_type(8)))  float  v8f;

// One packed bf16 pair per instruction: dst.l = bf16(a), dst.h = bf16(b).
__device__ __forceinline__ unsigned int pk_bf16(float a, float b) {
    unsigned int r;
    asm("v_cvt_pk_bf16_f32 %0, %1, %2" : "=v"(r) : "v"(a), "v"(b));
    return r;
}

// A fragment (16x32 bf16) from row-major bf16 LDS tile: 2x b128 per lane.
__device__ __forceinline__ v16bf a_frag_ld(const bf16_t* __restrict__ p,
                                           int stride, int m, int kbase,
                                           int lane) {
    const int khalf = (lane >> 4) << 3;
    const bf16_t* row = p + m * stride + kbase + khalf;
    union { v16bf v; uint4 q[2]; } u;
    u.q[0] = *(const uint4*)row;          // K = khalf .. khalf+7
    u.q[1] = *(const uint4*)(row + 16);   // K = 16+khalf .. 16+khalf+7
    return u.v;
}

// B fragment (32x16 bf16) from transposed [N][KSTR] LDS stage: 2x b128.
__device__ __forceinline__ v16bf b_frag_ld(const bf16_t* __restrict__ pT,
                                           int n, int lane) {
    const bf16_t* col = pT + (n + (lane & 15)) * KSTR + ((lane >> 4) << 3);
    union { v16bf v; uint4 q[2]; } u;
    u.q[0] = *(const uint4*)col;          // K = khalf .. khalf+7
    u.q[1] = *(const uint4*)(col + 16);   // K = 16+khalf .. 16+khalf+7
    return u.v;
}

__device__ __forceinline__ v8f wmma_bf16(v16bf a, v16bf b, v8f c) {
    return __builtin_amdgcn_wmma_f32_16x16x32_bf16(
        false, a, false, b, (short)0, c, false, false);
}

// ---- pipelined weight staging: global float4 pairs -> registers -----------
// GEMM1 chunk: 32 x 256 elems -> items (kpair 0..15, n4 0..63), 2 per thread
__device__ __forceinline__ void g1_load(const float* __restrict__ W1, int k0,
                                        int tid, float4 (&lo)[2], float4 (&hi)[2]) {
#pragma unroll
    for (int i = 0; i < 2; ++i) {
        int e = tid + NT * i;
        int kp = e >> 6, n4 = e & 63;
        const float* base = W1 + (size_t)(k0 + 2 * kp) * AD + (n4 << 2);
        lo[i] = *(const float4*)base;
        hi[i] = *(const float4*)(base + AD);
    }
}
__device__ __forceinline__ void g1_store(bf16_t* __restrict__ bT, int tid,
                                         const float4 (&lo)[2], const float4 (&hi)[2]) {
#pragma unroll
    for (int i = 0; i < 2; ++i) {
        int e = tid + NT * i;
        int kp = e >> 6, n4 = e & 63;
        const float* a = (const float*)&lo[i];
        const float* b = (const float*)&hi[i];
#pragma unroll
        for (int j = 0; j < 4; ++j) {
            *(unsigned int*)&bT[((n4 << 2) + j) * KSTR + 2 * kp] = pk_bf16(a[j], b[j]);
        }
    }
}
// GEMM2 chunk: 32 x 128 elems -> items (kpair 0..15, n4 0..31), 1 per thread
__device__ __forceinline__ void g2_load(const float* __restrict__ W2, int k0,
                                        int nbase, int tid,
                                        float4& lo, float4& hi) {
    int kp = tid >> 5, n4 = tid & 31;
    const float* base = W2 + (size_t)(k0 + 2 * kp) * HD + nbase + (n4 << 2);
    lo = *(const float4*)base;
    hi = *(const float4*)(base + HD);
}
__device__ __forceinline__ void g2_store(bf16_t* __restrict__ bT, int tid,
                                         const float4& lo, const float4& hi) {
    int kp = tid >> 5, n4 = tid & 31;
    const float* a = (const float*)&lo;
    const float* b = (const float*)&hi;
#pragma unroll
    for (int j = 0; j < 4; ++j) {
        *(unsigned int*)&bT[((n4 << 2) + j) * KSTR + 2 * kp] = pk_bf16(a[j], b[j]);
    }
}

__global__ __launch_bounds__(NT, 1)
void language_adapter_kernel(const float* __restrict__ hs,
                             const int*   __restrict__ lang_ids,
                             const float* __restrict__ W1g,
                             const float* __restrict__ b1g,
                             const float* __restrict__ W2g,
                             const float* __restrict__ b2g,
                             const float* __restrict__ gammag,
                             const float* __restrict__ betag,
                             float* __restrict__ out) {
    extern __shared__ char smem[];
    bf16_t* abf   = (bf16_t*)smem;                          // [BM][ASTR] hidden -> x
    bf16_t* htile = (bf16_t*)(smem + ABF_BYTES);            // [BM][HSTR] bottleneck
    bf16_t* bT0   = (bf16_t*)(smem + ABF_BYTES + HTILE_BYTES);
    bf16_t* bT1   = bT0 + BSTG_ELEMS;

    const int tid  = threadIdx.x;
    const int lane = tid & 31;
    const int wid  = tid >> 5;                 // 0..15
    const int b    = blockIdx.y;
    const int row0 = blockIdx.x * BM;
    const int lang = lang_ids[b];

    const float* W1 = W1g + (size_t)lang * HD * AD;
    const float* B1 = b1g + (size_t)lang * AD;
    const float* W2 = W2g + (size_t)lang * AD * HD;
    const float* B2 = b2g + (size_t)lang * HD;
    const float* GM = gammag + (size_t)lang * HD;
    const float* BT = betag  + (size_t)lang * HD;
    const float* hsrc = hs  + ((size_t)b * SEQ + row0) * HD;
    float*       dst  = out + ((size_t)b * SEQ + row0) * HD;

    // ---- hidden tile -> LDS (bf16): BM x HD --------------------------------
#pragma unroll 4
    for (int e = tid; e < BM * (HD / 4); e += NT) {
        int r  = e >> 8;                 // 256 float4 per row
        int c4 = (e & 255) << 2;
        float4 v = *(const float4*)&hsrc[r * HD + c4];
        uint2 p;
        p.x = pk_bf16(v.x, v.y);
        p.y = pk_bf16(v.z, v.w);
        *(uint2*)&abf[r * ASTR + c4] = p;
    }

    // prologue: first W1 chunk into registers
    float4 s1lo[2], s1hi[2];
    g1_load(W1, 0, tid, s1lo, s1hi);
    __syncthreads();

    // ---- GEMM1: h[BM][AD] = x @ W1 -----------------------------------------
    // waves: m0 = (wid&3)*16; 4 N-tiles at (wid>>2)*64 + t*16
    const int m0  = (wid & 3) << 4;
    const int nc0 = (wid >> 2) << 6;
    const int am  = m0 + (lane & 15);
    v8f acc1[4] = {};

#pragma unroll 2
    for (int kc = 0; kc < HD / BKK; ++kc) {
        bf16_t* buf = (kc & 1) ? bT1 : bT0;
        g1_store(buf, tid, s1lo, s1hi);
        __syncthreads();
        if (kc + 1 < HD / BKK) g1_load(W1, (kc + 1) * BKK, tid, s1lo, s1hi);

        v16bf a = a_frag_ld(abf, ASTR, am, kc * BKK, lane);
        // batch B-fragments in distinct regs: ds loads clause + pipelined waits
        v16bf b0 = b_frag_ld(buf, nc0 +  0, lane);
        v16bf b1 = b_frag_ld(buf, nc0 + 16, lane);
        v16bf b2 = b_frag_ld(buf, nc0 + 32, lane);
        v16bf b3 = b_frag_ld(buf, nc0 + 48, lane);
        acc1[0] = wmma_bf16(a, b0, acc1[0]);
        acc1[1] = wmma_bf16(a, b1, acc1[1]);
        acc1[2] = wmma_bf16(a, b2, acc1[2]);
        acc1[3] = wmma_bf16(a, b3, acc1[3]);
        // no trailing barrier: next chunk goes to the other LDS buffer
    }

    // prologue for GEMM2 (overlaps h-tile store)
    float4 s2lo, s2hi;
    g2_load(W2, 0, 0, tid, s2lo, s2hi);

    // bias + relu, store h tile (bf16, row-major)
#pragma unroll
    for (int t = 0; t < 4; ++t) {
        int n = nc0 + t * 16 + (lane & 15);
        float bias = B1[n];
#pragma unroll
        for (int i = 0; i < 8; ++i) {
            int m = m0 + ((lane >> 4) << 3) + i;
            float v = acc1[t][i] + bias;
            htile[m * HSTR + n] = (bf16_t)(v > 0.0f ? v : 0.0f);
        }
    }
    __syncthreads();   // htile visible; also closes out GEMM1 stage reads

    // ---- GEMM2: x = hidden + h @ W2 + b2, in 8 column chunks of 128 --------
    // waves: m0 as above; 2 N-tiles at (wid>>2)*32 + t*16
    const int nl0 = (wid >> 2) << 5;
    for (int nc = 0; nc < HD / 128; ++nc) {
        const int nbase = nc << 7;
        v8f acc2[2] = {};
#pragma unroll 2
        for (int kc = 0; kc < AD / BKK; ++kc) {
            bf16_t* buf = (kc & 1) ? bT1 : bT0;
            g2_store(buf, tid, s2lo, s2hi);
            __syncthreads();
            if (kc + 1 < AD / BKK)
                g2_load(W2, (kc + 1) * BKK, nbase, tid, s2lo, s2hi);
            else if (nc + 1 < HD / 128)
                g2_load(W2, 0, nbase + 128, tid, s2lo, s2hi);

            v16bf a = a_frag_ld(htile, HSTR, am, kc * BKK, lane);
            v16bf b0 = b_frag_ld(buf, nl0 +  0, lane);
            v16bf b1 = b_frag_ld(buf, nl0 + 16, lane);
            acc2[0] = wmma_bf16(a, b0, acc2[0]);
            acc2[1] = wmma_bf16(a, b1, acc2[1]);
            // no trailing barrier (double-buffered stage)
        }
        // residual: x = hidden + adapted + b2, back into abf (disjoint tiles)
#pragma unroll
        for (int t = 0; t < 2; ++t) {
            int ng = nbase + nl0 + t * 16 + (lane & 15);
            float bias = B2[ng];
#pragma unroll
            for (int i = 0; i < 8; ++i) {
                int m = m0 + ((lane >> 4) << 3) + i;
                float xv = (float)abf[m * ASTR + ng] + acc2[t][i] + bias;
                abf[m * ASTR + ng] = (bf16_t)xv;
            }
        }
    }
    __syncthreads();

    // ---- LayerNorm over H per row: one wave per row, 4 rows per wave -------
    for (int j = 0; j < BM / 16; ++j) {
        const int r = wid + (j << 4);
        const bf16_t* xr = &abf[r * ASTR];
        float xs[32];
        float s = 0.0f, s2 = 0.0f;
#pragma unroll
        for (int q = 0; q < HD / 256; ++q) {
            union { uint4 u; __bf16 h[8]; } v;
            v.u = *(const uint4*)&xr[(q * 32 + lane) << 3];
#pragma unroll
            for (int k = 0; k < 8; ++k) {
                float f = (float)v.h[k];
                xs[q * 8 + k] = f;
                s += f;
                s2 += f * f;
            }
        }
#pragma unroll
        for (int off = 16; off >= 1; off >>= 1) {
            s  += __shfl_xor(s, off, 32);
            s2 += __shfl_xor(s2, off, 32);
        }
        const float mean = s * (1.0f / HD);
        const float var  = s2 * (1.0f / HD) - mean * mean;
        const float rstd = rsqrtf(var + 1e-5f);
        float* orow = &dst[(size_t)r * HD];
#pragma unroll
        for (int q = 0; q < HD / 256; ++q) {
            int c = (q * 32 + lane) << 3;
            float4 g0 = *(const float4*)&GM[c];
            float4 g1 = *(const float4*)&GM[c + 4];
            float4 t0 = *(const float4*)&BT[c];
            float4 t1 = *(const float4*)&BT[c + 4];
            float4 o0, o1;
            o0.x = (xs[q*8+0] - mean) * rstd * g0.x + t0.x;
            o0.y = (xs[q*8+1] - mean) * rstd * g0.y + t0.y;
            o0.z = (xs[q*8+2] - mean) * rstd * g0.z + t0.z;
            o0.w = (xs[q*8+3] - mean) * rstd * g0.w + t0.w;
            o1.x = (xs[q*8+4] - mean) * rstd * g1.x + t1.x;
            o1.y = (xs[q*8+5] - mean) * rstd * g1.y + t1.y;
            o1.z = (xs[q*8+6] - mean) * rstd * g1.z + t1.z;
            o1.w = (xs[q*8+7] - mean) * rstd * g1.w + t1.w;
            *(float4*)&orow[c]     = o0;
            *(float4*)&orow[c + 4] = o1;
        }
    }
}

extern "C" void kernel_launch(void* const* d_in, const int* in_sizes, int n_in,
                              void* d_out, int out_size, void* d_ws, size_t ws_size,
                              hipStream_t stream) {
    (void)in_sizes; (void)n_in; (void)d_ws; (void)ws_size; (void)out_size;
    const float* hs    = (const float*)d_in[0];
    const int*   langs = (const int*)d_in[1];
    const float* W1    = (const float*)d_in[2];
    const float* b1    = (const float*)d_in[3];
    const float* W2    = (const float*)d_in[4];
    const float* b2    = (const float*)d_in[5];
    const float* gamma = (const float*)d_in[6];
    const float* beta  = (const float*)d_in[7];
    float* out = (float*)d_out;

    hipFuncSetAttribute((const void*)language_adapter_kernel,
                        hipFuncAttributeMaxDynamicSharedMemorySize, SMEM_BYTES);

    dim3 grid(SEQ / BM, BATCH);
    dim3 block(NT);
    language_adapter_kernel<<<grid, block, SMEM_BYTES, stream>>>(
        hs, langs, W1, b1, W2, b2, gamma, beta, out);
}
